// MRM_47244640256437
// MI455X (gfx1250) — matrix operand order repaired and checked by reading, not compiled, not run on hardware
//
#include <hip/hip_runtime.h>
#include <stdint.h>
#include <math.h>

// ---------------- model constants ----------------
#define TOKENS 1024      // 2 x 512
#define BATCH  2
#define SEQ    512
#define DMODEL 768
#define DINNER 1536      // 2*D
#define XZLD   3072      // 2*DINNER
#define EHID   3072      // expert hidden H
#define VOCAB  50257
#define HEADS  8
#define HD     96        // DMODEL/HEADS

typedef __attribute__((ext_vector_type(16))) __bf16 v16bf;
typedef __attribute__((ext_vector_type(8)))  __bf16 v8bf;
typedef __attribute__((ext_vector_type(8)))  float  v8f;

// ---------------- device helpers ----------------
__device__ __forceinline__ float wred_sum(float v){
#pragma unroll
  for (int o = 16; o > 0; o >>= 1) v += __shfl_xor(v, o, 32);
  return v;
}
__device__ __forceinline__ float wred_max(float v){
#pragma unroll
  for (int o = 16; o > 0; o >>= 1) v = fmaxf(v, __shfl_xor(v, o, 32));
  return v;
}
__device__ __forceinline__ float siluf(float x){ return x / (1.f + expf(-x)); }
__device__ __forceinline__ float geluf(float x){ return 0.5f * x * (1.f + erff(x * 0.70710678118f)); }
__device__ __forceinline__ float softplusf(float x){ return x > 20.f ? x : log1pf(expf(x)); }

// =================================================================
// Batched WMMA GEMM:  C[m,n] (+)= rowScale[m] * (act(alpha * sum_k A[m,k]*B(n,k) + bias[n]))
//   A: bf16 (M,K) row-major, lda.
//   TRANSB=1: B is bf16 (N,K) row-major, ldb (weight layout).
//   TRANSB=0: B is bf16 (K,N) row-major, ldb.
// One wave owns a 16x64 C tile: one A fragment feeds FOUR independent WMMAs.
// 8 waves/block -> 128x64 block tile.  K%32==0 (padded where needed).
// N-edge handling is branch-free: out-of-range n is CLAMPED to N-1 for loads
// (B lane n only feeds C column n, so garbage only lands in columns we skip
// at store time).
// DOUBLE-BUFFERED pipeline: fragments for chunk k+1 are loaded while the four
// WMMAs for chunk k execute, so each WMMA group only waits on its own chunk
// (partial loadcnt wait) instead of full memory latency per WMMA.
// Fragment layouts used here (contiguous k order, shared by A and B):
//   A lane l: m=l&15, half=l>>4 -> elem e    = K[half*16 + e]
//   B lane l: n=l&15, half=l>>4 -> elem e    = K[half*16 + e]
//   C lane l: vgpr r -> row tm + (l>>4)*8 + r, col tn + (l&15)
// =================================================================
template<int TRANSB>
__global__ void k_gemm_wmma(const unsigned short* __restrict__ Au,
                            const unsigned short* __restrict__ Bu,
                            float* __restrict__ C,
                            const float* __restrict__ bias,
                            const float* __restrict__ rowScale, int rsStride,
                            int N, int K, int lda, int ldb, int ldc,
                            float alpha, int act, int accum,
                            int batchInner,
                            long long sAo, long long sAi,
                            long long sBo, long long sBi,
                            long long sCo, long long sCi)
{
  const __bf16* A = reinterpret_cast<const __bf16*>(Au);
  const __bf16* B = reinterpret_cast<const __bf16*>(Bu);
  const int z  = blockIdx.z;
  const int zo = z / batchInner, zi = z - zo * batchInner;
  A += zo * sAo + zi * sAi;
  B += zo * sBo + zi * sBi;
  C += zo * sCo + zi * sCi;

  const int wave = threadIdx.x >> 5;
  const int lane = threadIdx.x & 31;
  const int half = lane >> 4;
  const int l15  = lane & 15;
  const int tm = blockIdx.y * 128 + wave * 16;
  const int tn = blockIdx.x * 64;
  const int m  = tm + l15;

  int  n[4];
  bool ok[4];
  const __bf16* bp[4];
#pragma unroll
  for (int t = 0; t < 4; ++t) {
    n[t]  = tn + t * 16 + l15;
    ok[t] = (n[t] < N);
    const int nc = ok[t] ? n[t] : (N - 1);       // clamp: loads stay in-bounds
    if (TRANSB) bp[t] = B + (size_t)nc * ldb + half * 16;
    else        bp[t] = B + (size_t)(half * 16) * ldb + nc;
  }

  v8f acc[4];
#pragma unroll
  for (int t = 0; t < 4; ++t) acc[t] = (v8f){0.f,0.f,0.f,0.f,0.f,0.f,0.f,0.f};

  const __bf16* arow = A + (size_t)m * lda + half * 16;

  // Load one 32-K chunk of fragments and advance the pointers.
  auto load_chunk = [&](v16bf& af, v16bf (&bf)[4]) {
    {
      v8bf x0 = *reinterpret_cast<const v8bf*>(arow);
      v8bf x1 = *reinterpret_cast<const v8bf*>(arow + 8);
#pragma unroll
      for (int e = 0; e < 8; ++e) { af[e] = x0[e]; af[8 + e] = x1[e]; }
    }
    arow += 32;
    if (TRANSB) {
#pragma unroll
      for (int t = 0; t < 4; ++t) {
        v8bf y0 = *reinterpret_cast<const v8bf*>(bp[t]);
        v8bf y1 = *reinterpret_cast<const v8bf*>(bp[t] + 8);
#pragma unroll
        for (int e = 0; e < 8; ++e) { bf[t][e] = y0[e]; bf[t][8 + e] = y1[e]; }
        bp[t] += 32;
      }
    } else {
#pragma unroll
      for (int t = 0; t < 4; ++t) {
#pragma unroll
        for (int e = 0; e < 16; ++e) bf[t][e] = bp[t][(size_t)e * ldb];
        bp[t] += (size_t)32 * ldb;
      }
    }
  };
  // EXEC is all-ones at the WMMAs; 4 independent accumulators.
  auto wmma4 = [&](const v16bf& af, const v16bf (&bf)[4]) {
#pragma unroll
    for (int t = 0; t < 4; ++t)
      acc[t] = __builtin_amdgcn_wmma_f32_16x16x32_bf16(false, af, false, bf[t], (short)0,
                                                       acc[t], false, false);
  };

  // -------- double-buffered main loop --------
  v16bf afA, afB;
  v16bf bfA[4], bfB[4];
  const int steps = K >> 5;          // K%32==0; steps in {1,2,3,16,24,48,96}
  load_chunk(afA, bfA);
  int i = 1;
  for (; i + 1 < steps; i += 2) {
    load_chunk(afB, bfB);            // prefetch chunk i
    wmma4(afA, bfA);                 // compute chunk i-1 (waits only on its loads)
    load_chunk(afA, bfA);            // prefetch chunk i+1
    wmma4(afB, bfB);                 // compute chunk i
  }
  if (i < steps) {                   // one more loaded pair (even tail)
    load_chunk(afB, bfB);
    wmma4(afA, bfA);
    wmma4(afB, bfB);
  } else {                           // odd tail
    wmma4(afA, bfA);
  }

  // -------- epilogue (4 tiles; store predicated on n<N only) --------
#pragma unroll
  for (int t = 0; t < 4; ++t) {
    if (!ok[t]) continue;
    const float bn = bias ? bias[n[t]] : 0.f;
#pragma unroll
    for (int r = 0; r < 8; ++r) {
      const int mr = tm + half * 8 + r;
      float v = alpha * acc[t][r] + bn;
      if (act == 1) v = softplusf(v);
      if (rowScale) v *= rowScale[(size_t)mr * rsStride];
      float* cp = C + (size_t)mr * ldc + n[t];
      if (accum) *cp += v; else *cp = v;
    }
  }
}

// f32 -> bf16 conversion with optional column padding; 8 elements per thread.
// Callers guarantee: total%8==0, srcLd%8==0, dstLd%8==0, copyCols%8==0.
__global__ void k_convert_pad8(unsigned short* __restrict__ dst, const float* __restrict__ src,
                               long long total8, int srcLd, int copyCols, int dstLd)
{
  long long i8 = (long long)blockIdx.x * blockDim.x + threadIdx.x;
  if (i8 >= total8) return;
  long long i = i8 * 8;
  int r = (int)(i / dstLd);
  int c = (int)(i - (long long)r * dstLd);
  unsigned int w[4];
  if (c < copyCols) {
    const float* sp = src + (size_t)r * srcLd + c;
    float4 f0 = *reinterpret_cast<const float4*>(sp);
    float4 f1 = *reinterpret_cast<const float4*>(sp + 4);
    float fv[8] = {f0.x, f0.y, f0.z, f0.w, f1.x, f1.y, f1.z, f1.w};
    unsigned short hv[8];
#pragma unroll
    for (int e = 0; e < 8; ++e) {
      __bf16 h = (__bf16)fv[e];
      hv[e] = *reinterpret_cast<unsigned short*>(&h);
    }
#pragma unroll
    for (int e = 0; e < 4; ++e) w[e] = (unsigned int)hv[2*e] | ((unsigned int)hv[2*e+1] << 16);
  } else {
#pragma unroll
    for (int e = 0; e < 4; ++e) w[e] = 0u;
  }
  uint4 out; out.x = w[0]; out.y = w[1]; out.z = w[2]; out.w = w[3];
  *reinterpret_cast<uint4*>(dst + i) = out;
}

__global__ void k_embed(float* __restrict__ X, const float* __restrict__ emb,
                        const int* __restrict__ ids)
{
  int i = blockIdx.x * blockDim.x + threadIdx.x;
  if (i >= TOKENS * DMODEL) return;
  int t = i / DMODEL, d = i - t * DMODEL;
  X[i] = emb[(size_t)ids[t] * DMODEL + d];
}

// depthwise causal conv (DCONV=4) over first DINNER cols of XZ, + bias, SiLU
__global__ void k_dwconv_silu(float* __restrict__ XI, const float* __restrict__ XZ,
                              const float* __restrict__ cw, const float* __restrict__ cb)
{
  int i = blockIdx.x * blockDim.x + threadIdx.x;
  if (i >= TOKENS * DINNER) return;
  int t = i / DINNER, c = i - t * DINNER;
  int b = t >> 9, s = t & 511;
  float acc = cb[c];
#pragma unroll
  for (int j = 0; j < 4; ++j) {
    int sp = s - 3 + j;
    if (sp >= 0) acc += cw[c * 4 + j] * XZ[(size_t)((b << 9) + sp) * XZLD + c];
  }
  XI[i] = siluf(acc);
}

// Mamba selective scan: one thread per (batch, channel), 8 states in registers.
// Fuses y = (scan + D*xi) * silu(z).
__global__ void k_scan(float* __restrict__ Y, const float* __restrict__ DT,
                       const float* __restrict__ XI, const float* __restrict__ DBL,
                       const float* __restrict__ XZ, const float* __restrict__ A_log,
                       const float* __restrict__ Dp)
{
  int idx = blockIdx.x * blockDim.x + threadIdx.x;
  if (idx >= BATCH * DINNER) return;
  int b = idx / DINNER, d = idx - b * DINNER;
  float Aa[8], h[8];
#pragma unroll
  for (int n = 0; n < 8; ++n) { Aa[n] = -expf(A_log[d * 8 + n]); h[n] = 0.f; }
  float Dv = Dp[d];
  for (int s = 0; s < SEQ; ++s) {
    int t = b * SEQ + s;
    float dt = DT[(size_t)t * DINNER + d];
    float xv = XI[(size_t)t * DINNER + d];
    const float* bl = DBL + (size_t)t * 64;   // cols 48..55 = B, 56..63 = C
    float y = 0.f;
#pragma unroll
    for (int n = 0; n < 8; ++n) {
      h[n] = expf(dt * Aa[n]) * h[n] + dt * bl[48 + n] * xv;
      y += h[n] * bl[56 + n];
    }
    y += Dv * xv;
    float z = XZ[(size_t)t * XZLD + DINNER + d];
    Y[(size_t)t * DINNER + d] = y * siluf(z);
  }
}

// GLU: a=first half, g=act(second half); kind 0=silu(conv expert), 1=gelu(ffn). Writes bf16.
__global__ void k_glu_bf16(unsigned short* __restrict__ GB, const float* __restrict__ G2, int kind)
{
  int i = blockIdx.x * blockDim.x + threadIdx.x;
  if (i >= TOKENS * EHID) return;
  int t = i / EHID, c = i - t * EHID;
  float a  = G2[(size_t)t * 2 * EHID + c];
  float bb = G2[(size_t)t * 2 * EHID + EHID + c];
  float g  = (kind == 0) ? siluf(bb) : geluf(bb);
  __bf16 h = (__bf16)(g * a);
  GB[i] = *reinterpret_cast<unsigned short*>(&h);
}

// ACC = wgt_id * XM  (identity expert initializes the MoE accumulator)
__global__ void k_moe_init(float* __restrict__ ACC, const float* __restrict__ XM,
                           const float* __restrict__ WGT, int E)
{
  int i = blockIdx.x * blockDim.x + threadIdx.x;
  if (i >= TOKENS * DMODEL) return;
  int t = i / DMODEL;
  ACC[i] = WGT[(size_t)t * E + (E - 1)] * XM[i];
}

// router logits + softmax + top-2 -> per-token expert weights (one wave per token)
__global__ void k_router(float* __restrict__ WGT, const float* __restrict__ XM,
                         const float* __restrict__ RW, const float* __restrict__ RB, int E)
{
  int wv = threadIdx.x >> 5, lane = threadIdx.x & 31;
  int t = blockIdx.x * 8 + wv;
  if (t >= TOKENS) return;
  float acc[7];
#pragma unroll
  for (int e = 0; e < 7; ++e) acc[e] = 0.f;
  const float* x = XM + (size_t)t * DMODEL;
  for (int d = lane; d < DMODEL; d += 32) {
    float xv = x[d];
    for (int e = 0; e < E; ++e) acc[e] += xv * RW[(size_t)e * DMODEL + d];
  }
  for (int e = 0; e < E; ++e) acc[e] = wred_sum(acc[e]);
  if (lane == 0) {
    float p[7]; float mx = -3.4e38f;
    for (int e = 0; e < E; ++e) { p[e] = acc[e] + RB[e]; mx = fmaxf(mx, p[e]); }
    float s = 0.f;
    for (int e = 0; e < E; ++e) { p[e] = expf(p[e] - mx); s += p[e]; }
    for (int e = 0; e < E; ++e) p[e] /= s;
    int i1 = 0;
    for (int e = 1; e < E; ++e) if (p[e] > p[i1]) i1 = e;
    int i2 = (i1 == 0) ? 1 : 0;
    for (int e = 0; e < E; ++e) if (e != i1 && p[e] > p[i2]) i2 = e;
    for (int e = 0; e < E; ++e)
      WGT[(size_t)t * E + e] = (e == i1) ? p[i1] : ((e == i2) ? p[i2] : 0.f);
  }
}

// dst = LayerNorm(A + B) * g + beta  (one wave per token; dst may alias A or B)
__global__ void k_ln_res(float* __restrict__ dst, const float* __restrict__ A,
                         const float* __restrict__ Bs, const float* __restrict__ g,
                         const float* __restrict__ beta)
{
  int wv = threadIdx.x >> 5, lane = threadIdx.x & 31;
  int t = blockIdx.x * 8 + wv;
  if (t >= TOKENS) return;
  const float* pa = A  + (size_t)t * DMODEL;
  const float* pb = Bs + (size_t)t * DMODEL;
  float s = 0.f, sq = 0.f;
  for (int d = lane; d < DMODEL; d += 32) { float v = pa[d] + pb[d]; s += v; sq += v * v; }
  s = wred_sum(s); sq = wred_sum(sq);
  float mean = s * (1.f / DMODEL);
  float var  = sq * (1.f / DMODEL) - mean * mean;
  float inv  = rsqrtf(var + 1e-5f);
  float* pd = dst + (size_t)t * DMODEL;
  for (int d = lane; d < DMODEL; d += 32) {
    float v = pa[d] + pb[d];
    pd[d] = (v - mean) * inv * g[d] + beta[d];
  }
}

// row softmax (f32 in) -> bf16 out (one wave per row)
__global__ void k_softmax_bf16(unsigned short* __restrict__ dst, const float* __restrict__ src,
                               int rows, int cols)
{
  int wv = threadIdx.x >> 5, lane = threadIdx.x & 31;
  int row = blockIdx.x * 8 + wv;
  if (row >= rows) return;
  const float* p = src + (size_t)row * cols;
  float mx = -3.4e38f;
  for (int c = lane; c < cols; c += 32) mx = fmaxf(mx, p[c]);
  mx = wred_max(mx);
  float s = 0.f;
  for (int c = lane; c < cols; c += 32) s += expf(p[c] - mx);
  s = wred_sum(s);
  float inv = 1.f / s;
  unsigned short* q = dst + (size_t)row * cols;
  for (int c = lane; c < cols; c += 32) {
    __bf16 h = (__bf16)(expf(p[c] - mx) * inv);
    q[c] = *reinterpret_cast<unsigned short*>(&h);
  }
}

// ---------------- host helpers ----------------
static inline int cdiv(int a, int b){ return (a + b - 1) / b; }

static void launch_gemm(hipStream_t st, const unsigned short* A, const unsigned short* B, float* C,
                        const float* bias, const float* rowScale, int rsStride,
                        int M, int N, int K, int lda, int ldb, int ldc,
                        float alpha, int act, int accum, int transB,
                        int bo, int bi,
                        long long sAo, long long sAi, long long sBo, long long sBi,
                        long long sCo, long long sCi)
{
  dim3 grid(cdiv(N, 64), M / 128, bo * bi);
  if (transB)
    k_gemm_wmma<1><<<grid, 256, 0, st>>>(A, B, C, bias, rowScale, rsStride,
                                         N, K, lda, ldb, ldc, alpha, act, accum,
                                         bi, sAo, sAi, sBo, sBi, sCo, sCi);
  else
    k_gemm_wmma<0><<<grid, 256, 0, st>>>(A, B, C, bias, rowScale, rsStride,
                                         N, K, lda, ldb, ldc, alpha, act, accum,
                                         bi, sAo, sAi, sBo, sBi, sCo, sCi);
}
static void gemmN(hipStream_t st, const unsigned short* A, const unsigned short* B, float* C,
                  const float* bias, const float* rowScale, int rsStride,
                  int M, int N, int K, int lda, int ldb, int ldc,
                  float alpha, int act, int accum)
{
  launch_gemm(st, A, B, C, bias, rowScale, rsStride, M, N, K, lda, ldb, ldc,
              alpha, act, accum, /*transB=*/1, 1, 1, 0, 0, 0, 0, 0, 0);
}
static void conv_bf16(hipStream_t st, unsigned short* dst, const float* src,
                      long long rows, int srcLd, int copyCols, int dstLd)
{
  long long total8 = rows * (long long)dstLd / 8;
  int blocks = (int)((total8 + 255) / 256);
  k_convert_pad8<<<blocks, 256, 0, st>>>(dst, src, total8, srcLd, copyCols, dstLd);
}

// =================================================================
// kernel_launch: full forward pass.
// Input leaf order follows jax.tree_util flattening (dict keys sorted
// alphabetically, lists in order):
//   input_ids, embed, lm_head, then per stage: blocks[] then experts[].
//   non-attn block: ln_b, ln_g, mamba{A_log,D,conv_b,conv_w,dt_b,dt_w,in_w,out_w,x_proj_w}, router_b, router_w
//   attn block: attn_in_b, attn_in_w, attn_out_b, attn_out_w, ln2_b, ln2_g, <non-attn leaves>
//   expert: b1, b2, w1, w2
// =================================================================
extern "C" void kernel_launch(void* const* d_in, const int* in_sizes, int n_in,
                              void* d_out, int out_size, void* d_ws, size_t ws_size,
                              hipStream_t stream)
{
  (void)in_sizes; (void)n_in; (void)out_size; (void)ws_size;

  int p = 0;
  const int*   ids     = (const int*)d_in[p++];
  const float* embedW  = (const float*)d_in[p++];
  const float* lm_head = (const float*)d_in[p++];

  const int  nbArr[3]   = {4, 6, 2};
  const int  nConv[3]   = {4, 3, 1};
  const int  nFfn[3]    = {2, 3, 4};
  const bool attnArr[3] = {false, false, true};

  struct MambaW { const float *A_log,*D,*conv_b,*conv_w,*dt_b,*dt_w,*in_w,*out_w,*x_proj_w; };
  struct BlockW {
    const float *attn_in_b,*attn_in_w,*attn_out_b,*attn_out_w,*ln2_b,*ln2_g;
    const float *ln_b,*ln_g; MambaW m; const float *router_b,*router_w;
  };
  struct ExpertW { const float *b1,*b2,*w1,*w2; };

  BlockW  blocks[3][6];
  ExpertW experts[3][6];
  for (int s = 0; s < 3; ++s) {
    for (int k = 0; k < nbArr[s]; ++k) {
      BlockW& B = blocks[s][k];
      if (attnArr[s]) {
        B.attn_in_b  = (const float*)d_in[p++];
        B.attn_in_w  = (const float*)d_in[p++];
        B.attn_out_b = (const float*)d_in[p++];
        B.attn_out_w = (const float*)d_in[p++];
        B.ln2_b      = (const float*)d_in[p++];
        B.ln2_g      = (const float*)d_in[p++];
      }
      B.ln_b = (const float*)d_in[p++];
      B.ln_g = (const float*)d_in[p++];
      B.m.A_log    = (const float*)d_in[p++];
      B.m.D        = (const float*)d_in[p++];
      B.m.conv_b   = (const float*)d_in[p++];
      B.m.conv_w   = (const float*)d_in[p++];
      B.m.dt_b     = (const float*)d_in[p++];
      B.m.dt_w     = (const float*)d_in[p++];
      B.m.in_w     = (const float*)d_in[p++];
      B.m.out_w    = (const float*)d_in[p++];
      B.m.x_proj_w = (const float*)d_in[p++];
      B.router_b = (const float*)d_in[p++];
      B.router_w = (const float*)d_in[p++];
    }
    int ne = nConv[s] + nFfn[s];
    for (int e = 0; e < ne; ++e) {
      experts[s][e].b1 = (const float*)d_in[p++];
      experts[s][e].b2 = (const float*)d_in[p++];
      experts[s][e].w1 = (const float*)d_in[p++];
      experts[s][e].w2 = (const float*)d_in[p++];
    }
  }

  // ---- workspace carve (deterministic, 256B aligned) ----
  char* wsb = (char*)d_ws;
  size_t off = 0;
  auto carve = [&](size_t bytes) -> char* {
    char* r = wsb + off;
    off = (off + bytes + 255) & ~(size_t)255;
    return r;
  };
  float* X    = (float*)carve((size_t)TOKENS * DMODEL * 4);   // residual / current activations
  float* XM   = (float*)carve((size_t)TOKENS * DMODEL * 4);   // mamba out / MoE input
  float* ACC  = (float*)carve((size_t)TOKENS * DMODEL * 4);   // MoE accumulator
  float* XZ   = (float*)carve((size_t)TOKENS * XZLD   * 4);   // in_proj out; reused as QKV f32
  float* XI   = (float*)carve((size_t)TOKENS * DINNER * 4);
  float* DTb  = (float*)carve((size_t)TOKENS * DINNER * 4);
  float* DBL  = (float*)carve((size_t)TOKENS * 64     * 4);
  float* Y    = (float*)carve((size_t)TOKENS * DINNER * 4);   // scan out; reused: O | AO
  float* WGT  = (float*)carve((size_t)TOKENS * 8      * 4);
  float* G2   = (float*)carve((size_t)TOKENS * 2 * EHID * 4); // expert h1; reused as attn scores f32
  unsigned short* ABUF = (unsigned short*)carve((size_t)TOKENS * EHID   * 2); // general bf16 acts / QKV bf16
  unsigned short* XB   = (unsigned short*)carve((size_t)TOKENS * DMODEL * 2); // persistent bf16 (MoE in / misc)
  unsigned short* SCB  = (unsigned short*)carve((size_t)16 * 512 * 512 * 2);  // attn probs bf16
  unsigned short* WBUF = (unsigned short*)carve((size_t)VOCAB * DMODEL * 2);  // weight bf16 staging

  auto ewg = [&](long long total) { return dim3((unsigned)((total + 255) / 256)); };

  // ---- embedding ----
  k_embed<<<ewg((long long)TOKENS * DMODEL), 256, 0, stream>>>(X, embedW, ids);

  for (int s = 0; s < 3; ++s) {
    for (int k = 0; k < nbArr[s]; ++k) {
      const BlockW& B = blocks[s][k];

      // ================= Mamba =================
      conv_bf16(stream, ABUF, X, TOKENS, DMODEL, DMODEL, DMODEL);
      conv_bf16(stream, WBUF, B.m.in_w, 2 * DINNER, DMODEL, DMODEL, DMODEL);
      gemmN(stream, ABUF, WBUF, XZ, nullptr, nullptr, 0,
            TOKENS, 2 * DINNER, DMODEL, DMODEL, DMODEL, XZLD, 1.f, 0, 0);

      k_dwconv_silu<<<ewg((long long)TOKENS * DINNER), 256, 0, stream>>>(XI, XZ, B.m.conv_w, B.m.conv_b);

      conv_bf16(stream, ABUF, XI, TOKENS, DINNER, DINNER, DINNER);
      conv_bf16(stream, WBUF, B.m.x_proj_w, 64, DINNER, DINNER, DINNER);
      gemmN(stream, ABUF, WBUF, DBL, nullptr, nullptr, 0,
            TOKENS, 64, DINNER, DINNER, DINNER, 64, 1.f, 0, 0);

      // dt = softplus(dbl[:, :48] @ dt_w.T + dt_b)   (K padded 48 -> 64)
      conv_bf16(stream, ABUF, DBL, TOKENS, 64, 48, 64);
      conv_bf16(stream, WBUF, B.m.dt_w, DINNER, 48, 48, 64);
      gemmN(stream, ABUF, WBUF, DTb, B.m.dt_b, nullptr, 0,
            TOKENS, DINNER, 64, 64, 64, DINNER, 1.f, /*softplus*/1, 0);

      k_scan<<<cdiv(BATCH * DINNER, 128), 128, 0, stream>>>(Y, DTb, XI, DBL, XZ, B.m.A_log, B.m.D);

      conv_bf16(stream, ABUF, Y, TOKENS, DINNER, DINNER, DINNER);
      conv_bf16(stream, WBUF, B.m.out_w, DMODEL, DINNER, DINNER, DINNER);
      gemmN(stream, ABUF, WBUF, XM, nullptr, nullptr, 0,
            TOKENS, DMODEL, DINNER, DINNER, DINNER, DMODEL, 1.f, 0, 0);

      // ================= MoE =================
      const int E = nConv[s] + nFfn[s] + 1;
      k_router<<<TOKENS / 8, 256, 0, stream>>>(WGT, XM, B.router_w, B.router_b, E);
      k_moe_init<<<ewg((long long)TOKENS * DMODEL), 256, 0, stream>>>(ACC, XM, WGT, E);
      conv_bf16(stream, XB, XM, TOKENS, DMODEL, DMODEL, DMODEL);
      for (int e = 0; e < E - 1; ++e) {
        const ExpertW& Ex = experts[s][e];
        conv_bf16(stream, WBUF, Ex.w1, 2 * EHID, DMODEL, DMODEL, DMODEL);
        gemmN(stream, XB, WBUF, G2, Ex.b1, nullptr, 0,
              TOKENS, 2 * EHID, DMODEL, DMODEL, DMODEL, 2 * EHID, 1.f, 0, 0);
        k_glu_bf16<<<ewg((long long)TOKENS * EHID), 256, 0, stream>>>(ABUF, G2, (e < nConv[s]) ? 0 : 1);
        conv_bf16(stream, WBUF, Ex.w2, DMODEL, EHID, EHID, EHID);
        // fused: ACC += wgt[:,e] * (g @ w2.T + b2)
        gemmN(stream, ABUF, WBUF, ACC, Ex.b2, WGT + e, E,
              TOKENS, DMODEL, EHID, EHID, EHID, DMODEL, 1.f, 0, /*accum*/1);
      }

      // ================= Attention (stage 2) =================
      if (attnArr[s]) {
        conv_bf16(stream, XB, ACC, TOKENS, DMODEL, DMODEL, DMODEL);
        conv_bf16(stream, WBUF, B.attn_in_w, 3 * DMODEL, DMODEL, DMODEL, DMODEL);
        gemmN(stream, XB, WBUF, XZ /*QKV*/, B.attn_in_b, nullptr, 0,
              TOKENS, 3 * DMODEL, DMODEL, DMODEL, DMODEL, 3 * DMODEL, 1.f, 0, 0);
        conv_bf16(stream, ABUF, XZ, TOKENS, 3 * DMODEL, 3 * DMODEL, 3 * DMODEL); // QKV bf16

        // scores[b,h] = (q @ k^T) / sqrt(hd)   (batched over 2x8)
        launch_gemm(stream, ABUF, ABUF + DMODEL, G2 /*SC*/, nullptr, nullptr, 0,
                    SEQ, SEQ, HD, 3 * DMODEL, 3 * DMODEL, SEQ,
                    1.f / sqrtf((float)HD), 0, 0, /*transB=*/1,
                    BATCH, HEADS,
                    (long long)SEQ * 3 * DMODEL, HD,
                    (long long)SEQ * 3 * DMODEL, HD,
                    (long long)HEADS * SEQ * SEQ, (long long)SEQ * SEQ);

        k_softmax_bf16<<<(16 * SEQ) / 8, 256, 0, stream>>>(SCB, G2, 16 * SEQ, SEQ);

        // o[b,h] = probs @ v   (B not transposed), written as (b,s,h,hd)
        launch_gemm(stream, SCB, ABUF + 2 * DMODEL, Y /*O*/, nullptr, nullptr, 0,
                    SEQ, HD, SEQ, SEQ, 3 * DMODEL, DMODEL,
                    1.f, 0, 0, /*transB=*/0,
                    BATCH, HEADS,
                    (long long)HEADS * SEQ * SEQ, (long long)SEQ * SEQ,
                    (long long)SEQ * 3 * DMODEL, HD,
                    (long long)SEQ * DMODEL, HD);

        conv_bf16(stream, XB, Y /*O*/, TOKENS, DMODEL, DMODEL, DMODEL);
        conv_bf16(stream, WBUF, B.attn_out_w, DMODEL, DMODEL, DMODEL, DMODEL);
        float* AO = Y + (size_t)TOKENS * DMODEL;
        gemmN(stream, XB, WBUF, AO, B.attn_out_b, nullptr, 0,
              TOKENS, DMODEL, DMODEL, DMODEL, DMODEL, DMODEL, 1.f, 0, 0);

        k_ln_res<<<TOKENS / 8, 256, 0, stream>>>(ACC, ACC, AO, B.ln2_g, B.ln2_b);
      }

      // x = LN(moe_out + residual)
      k_ln_res<<<TOKENS / 8, 256, 0, stream>>>(X, ACC, X, B.ln_g, B.ln_b);
    }
  }

  // ---- lm head: (1024, 50257) f32 out ----
  conv_bf16(stream, XB, X, TOKENS, DMODEL, DMODEL, DMODEL);
  conv_bf16(stream, WBUF, lm_head, VOCAB, DMODEL, DMODEL, DMODEL);
  gemmN(stream, XB, WBUF, (float*)d_out, nullptr, nullptr, 0,
        TOKENS, VOCAB, DMODEL, DMODEL, DMODEL, VOCAB, 1.f, 0, 0);
}
